// Upsampling_1872605741500
// MI455X (gfx1250) — matrix operand order repaired and checked by reading, not compiled, not run on hardware
//
#include <hip/hip_runtime.h>
#include <hip/hip_bf16.h>

// Problem constants (match reference)
#define BB 32
#define TT 512     // T_TEXT (K of the GEMM)
#define DD 512     // D     (N of the GEMM)
#define FF 4096    // T_FRAMES (M of the GEMM)

#define NEG_HALF_LOG_2PI (-0.9189385332046727f)

typedef _Float16 h8  __attribute__((ext_vector_type(8)));
typedef _Float16 v16h __attribute__((ext_vector_type(16)));
typedef float    v8f  __attribute__((ext_vector_type(8)));

union Frag16 { v16h v; h8 h[2]; };

// ---------------------------------------------------------------------------
// Kernel 0: per-token prep. One thread per batch row (B=32): serial cumsum of
// duration -> center; 1/sigma; logcoef = -log(sigma) - 0.5*log(2pi), with the
// text mask folded in as -inf so exp() -> 0 exactly.
// ---------------------------------------------------------------------------
__global__ void prep_tokens_kernel(const int* __restrict__ duration,
                                   const float* __restrict__ sigma,
                                   const unsigned char* __restrict__ mask,
                                   float* __restrict__ center,
                                   float* __restrict__ invsig,
                                   float* __restrict__ logcoef) {
  int b = threadIdx.x;
  if (b >= BB) return;
  int run = 0;
  const int base = b * TT;
  for (int t = 0; t < TT; ++t) {
    int d = duration[base + t];
    run += d;
    center[base + t] = (float)(run - (d >> 1));     // duration in [0,16)
    float sg = sigma[base + t];
    invsig[base + t] = 1.0f / sg;
    float lg = -__logf(sg) + NEG_HALF_LOG_2PI;
    if (mask[base + t]) lg = -__builtin_inff();     // exp(-inf) == 0
    logcoef[base + t] = lg;
  }
}

// ---------------------------------------------------------------------------
// Kernel 1: memory [b][t][d] f32  ->  memT [b][d][t] f16.
// Makes WMMA B-fragments contiguous 16B loads along K (=t). 16 MB: L2-resident.
// ---------------------------------------------------------------------------
__global__ void transpose_mem_kernel(const float* __restrict__ memory,
                                     _Float16* __restrict__ memT) {
  size_t i = (size_t)blockIdx.x * blockDim.x + threadIdx.x;   // B*T*D = 2^23
  int d = (int)(i & (DD - 1));
  int t = (int)((i >> 9) & (TT - 1));
  int b = (int)(i >> 18);
  memT[((size_t)(b * DD + d)) * TT + t] = (_Float16)memory[i];
}

// ---------------------------------------------------------------------------
// Kernel 2: per-(b,f) sum over t of exp(-0.5 z^2 + logcoef). Token params
// staged in LDS (6 KB). Writes 1/(sum + 1e-8).
// ---------------------------------------------------------------------------
__global__ void row_sums_kernel(const float* __restrict__ center,
                                const float* __restrict__ invsig,
                                const float* __restrict__ logcoef,
                                float* __restrict__ invsum) {
  __shared__ float sC[TT], sI[TT], sL[TT];
  const int b = blockIdx.x >> 4;                 // 16 blocks of 256 frames / batch
  const int f = ((blockIdx.x & 15) << 8) + threadIdx.x;
  for (int t = threadIdx.x; t < TT; t += 256) {
    sC[t] = center[b * TT + t];
    sI[t] = invsig[b * TT + t];
    sL[t] = logcoef[b * TT + t];
  }
  __syncthreads();
  const float fr = (float)f;
  float s = 0.0f;
#pragma unroll 4
  for (int t = 0; t < TT; ++t) {
    float z = (fr - sC[t]) * sI[t];
    s += __expf(fmaf(-0.5f * z, z, sL[t]));
  }
  invsum[b * FF + f] = 1.0f / (s + 1e-8f);
}

// ---------------------------------------------------------------------------
// Kernel 3: write normalized attn [b][t][f] (f contiguous -> coalesced 256 MB).
// Recomputes exp instead of reading back unnormalized weights (saves 512 MB of
// HBM traffic vs. a store+reload normalize pass).
// ---------------------------------------------------------------------------
__global__ void write_attn_kernel(const float* __restrict__ center,
                                  const float* __restrict__ invsig,
                                  const float* __restrict__ logcoef,
                                  const float* __restrict__ invsum,
                                  float* __restrict__ attn) {
  size_t i = (size_t)blockIdx.x * blockDim.x + threadIdx.x;   // B*T*F = 2^26
  int f = (int)(i & (FF - 1));
  int t = (int)((i >> 12) & (TT - 1));
  int b = (int)(i >> 21);
  float z = ((float)f - center[b * TT + t]) * invsig[b * TT + t];
  float w = __expf(fmaf(-0.5f * z, z, logcoef[b * TT + t]));
  attn[i] = w * invsum[b * FF + f];
}

// ---------------------------------------------------------------------------
// Kernel 4: batched GEMM  upsampled[b,f,d] = sum_t attn[b,t,f] * memory[b,t,d]
// via v_wmma_f32_16x16x32_f16. One block (8 waves) per (b, 16-row f-strip):
//   - A-tile (16 f x 512 t) converted f32->f16 cooperatively into LDS,
//     transposing [t][f] -> [f][t] so A fragments are ds_load_b128 pairs.
//   - Each wave owns 4 contiguous 16-wide d-tiles; B fragments are two
//     contiguous 16B global loads from the f16-transposed memT (L2-resident).
//   - 16 k-steps x 4 tiles = 64 WMMA per wave, f32 accumulators.
// A fragment lane layout per ISA (16-bit A 16x32): lanes 0-15 row M=lane,
// K = {0..7, 16..23}; lanes 16-31 row M=lane-16, K = {8..15, 24..31}.
// B fragment (32x16): lanes 0-15 col N=lane hold K = {0..7, 16..23}; lanes 16-31 K = {8..15, 24..31}.
// ---------------------------------------------------------------------------
#define LDA (TT + 8)   // pad 16B to stagger LDS banks

__global__ void __launch_bounds__(256)
upsample_wmma_kernel(const float* __restrict__ attn,
                     const _Float16* __restrict__ memT,
                     float* __restrict__ out) {
  __shared__ __align__(16) _Float16 Atile[16 * LDA];

  const int b  = blockIdx.x >> 8;          // 256 f-strips per batch
  const int f0 = (blockIdx.x & 255) << 4;
  const int tid = threadIdx.x;

  // Cooperative A-tile load+convert+transpose: attn[b][t][f0+j] -> Atile[j][t]
  const float* abase = attn + (size_t)b * TT * FF + f0;
#pragma unroll
  for (int c = 0; c < 32; ++c) {
    int idx = c * 256 + tid;               // 0..8191
    int t = idx >> 4;
    int j = idx & 15;
    Atile[j * LDA + t] = (_Float16)abase[(size_t)t * FF + j];
  }
  __syncthreads();

  const int wave  = tid >> 5;
  const int lane  = tid & 31;
  const int m     = lane & 15;             // row (A) / col (B) within tile
  const int khalf = lane >> 4;             // lane-group selector

  v8f acc[4];
#pragma unroll
  for (int j = 0; j < 4; ++j) acc[j] = v8f{};

  const _Float16* bbase = memT + (size_t)b * DD * TT;

  for (int kk = 0; kk < 16; ++kk) {
    const int kbA0 = kk * 32 + khalf * 8;  // A: {0..7}/{8..15} of this k-slice
    const int kbA1 = kbA0 + 16;            //    {16..23}/{24..31}
    const int kbB  = kk * 32 + khalf * 8;  // B: same K order as A

    Frag16 afr;
    afr.h[0] = *(const h8*)&Atile[m * LDA + kbA0];   // ds_load_b128
    afr.h[1] = *(const h8*)&Atile[m * LDA + kbA1];

#pragma unroll
    for (int j = 0; j < 4; ++j) {
      const _Float16* bp = bbase + (size_t)((wave * 4 + j) * 16 + m) * TT;
      Frag16 bfr;
      bfr.h[0] = *(const h8*)(bp + kbB);             // global_load_b128
      bfr.h[1] = *(const h8*)(bp + kbB + 16);
      if (kk < 15) __builtin_prefetch(bp + kbB + 32, 0, 1);  // next k-slice
      acc[j] = __builtin_amdgcn_wmma_f32_16x16x32_f16(
          /*neg_a=*/false, afr.v, /*neg_b=*/false, bfr.v,
          /*c_mod=*/(short)0, acc[j], /*reuse_a=*/false, /*reuse_b=*/false);
    }
  }

  // Store: C/D layout — VGPR r holds row M = r + 8*khalf, col N = m.
  float* obase = out + ((size_t)b * FF + f0) * DD;
#pragma unroll
  for (int j = 0; j < 4; ++j) {
    const int d0 = (wave * 4 + j) * 16;
#pragma unroll
    for (int r = 0; r < 8; ++r) {
      obase[(size_t)(r + 8 * khalf) * DD + d0 + m] = acc[j][r];
    }
  }
}

// ---------------------------------------------------------------------------
// Host launcher
// ---------------------------------------------------------------------------
extern "C" void kernel_launch(void* const* d_in, const int* in_sizes, int n_in,
                              void* d_out, int out_size, void* d_ws, size_t ws_size,
                              hipStream_t stream) {
  (void)in_sizes; (void)n_in; (void)out_size; (void)ws_size;

  // Inputs (setup_inputs order): memory f32, duration i32, sigma f32,
  // output_lengths i32 (unused; T_FRAMES is static), mask bool(u8)
  const float*         memory   = (const float*)d_in[0];
  const int*           duration = (const int*)d_in[1];
  const float*         sigma    = (const float*)d_in[2];
  const unsigned char* mask     = (const unsigned char*)d_in[4];

  // Outputs concatenated: upsampled [B,F,D] then attn [B,T,F]
  float* upsampled = (float*)d_out;
  float* attn      = (float*)d_out + (size_t)BB * FF * DD;

  // Workspace layout
  float* center  = (float*)d_ws;                        // B*T
  float* invsig  = center + BB * TT;                    // B*T
  float* logcoef = invsig + BB * TT;                    // B*T
  float* invsum  = logcoef + BB * TT;                   // B*F
  _Float16* memT = (_Float16*)((char*)d_ws + 720896);   // 16B-aligned, B*D*T f16

  prep_tokens_kernel<<<1, 32, 0, stream>>>(duration, sigma, mask,
                                           center, invsig, logcoef);

  transpose_mem_kernel<<<(BB * TT * DD) / 256, 256, 0, stream>>>(memory, memT);

  row_sums_kernel<<<BB * (FF / 256), 256, 0, stream>>>(center, invsig, logcoef,
                                                       invsum);

  write_attn_kernel<<<(size_t)(BB) * TT * FF / 256, 256, 0, stream>>>(
      center, invsig, logcoef, invsum, attn);

  upsample_wmma_kernel<<<BB * (FF / 16), 256, 0, stream>>>(attn, memT,
                                                           upsampled);
}
